// SelfAttention_26620207300861
// MI455X (gfx1250) — matrix operand order repaired and checked by reading, not compiled, not run on hardware
//
#include <hip/hip_runtime.h>
#include <hip/hip_bf16.h>
#include <math.h>

typedef __attribute__((ext_vector_type(16))) __bf16 v16bf;
typedef __attribute__((ext_vector_type(8)))  float  v8f;

#define IDIM 1024
#define DQK  512
#define DV   512
#define BATCH 4
#define SEQ  2048
#define NROW (BATCH*SEQ)   // 8192

// round-to-nearest-even f32 -> bf16 bit pattern
__device__ __forceinline__ unsigned short f2bf(float f) {
    unsigned int u = __float_as_uint(f);
    unsigned int r = (u + 0x7FFFu + ((u >> 16) & 1u)) >> 16;
    return (unsigned short)r;
}

// Build a 16x bf16 fragment from two contiguous 16-byte chunks.
__device__ __forceinline__ v16bf ld_frag(const unsigned short* p0, const unsigned short* p1) {
    v16bf v;
    *reinterpret_cast<float4*>(&v)       = *reinterpret_cast<const float4*>(p0);
    *(reinterpret_cast<float4*>(&v) + 1) = *reinterpret_cast<const float4*>(p1);
    return v;
}

// ---------------- prep: x -> bf16 ----------------
__global__ void k_cvt_x(const float* __restrict__ x, unsigned short* __restrict__ xb, int n) {
    int i = blockIdx.x * blockDim.x + threadIdx.x;
    if (i < n) xb[i] = f2bf(x[i]);
}

// ---------------- prep: W (K x N, f32) -> Wt (N x K, bf16) ----------------
__global__ void k_tr_w(const float* __restrict__ Wq, const float* __restrict__ Wk,
                       const float* __restrict__ Wv,
                       unsigned short* __restrict__ Wtq, unsigned short* __restrict__ Wtk,
                       unsigned short* __restrict__ Wtv) {
    int i = blockIdx.x * blockDim.x + threadIdx.x;      // over 512*1024
    const float* src;
    unsigned short* dst;
    if (blockIdx.y == 0)      { src = Wq; dst = Wtq; }
    else if (blockIdx.y == 1) { src = Wk; dst = Wtk; }
    else                      { src = Wv; dst = Wtv; }
    int k = i & (IDIM - 1);
    int n = i >> 10;
    dst[i] = f2bf(src[k * DQK + n]);   // Wt[n][k] = W[k][n]
}

// ---------------- QKV projection GEMM (bf16 WMMA, f32 accum) ----------------
// z=0: Q (scale+bias folded, row-major)   z=1: K (row-major)   z=2: V stored transposed [dv][s]
__global__ __launch_bounds__(256) void k_proj(
    const unsigned short* __restrict__ xb,
    const unsigned short* __restrict__ Wtq, const unsigned short* __restrict__ Wtk,
    const unsigned short* __restrict__ Wtv,
    const float* __restrict__ bq, const float* __restrict__ bk, const float* __restrict__ bv,
    unsigned short* __restrict__ Qs, unsigned short* __restrict__ Ks,
    unsigned short* __restrict__ Vt) {

    const int z = blockIdx.z;
    const unsigned short* Wt = (z == 0) ? Wtq : (z == 1) ? Wtk : Wtv;
    const float* bias        = (z == 0) ? bq  : (z == 1) ? bk  : bv;

    const int lane = threadIdx.x & 31;
    const int wave = threadIdx.x >> 5;
    const int mrow = lane & 15;
    const int half = lane >> 4;
    const int wr0  = blockIdx.y * 128 + wave * 16;   // 16 output rows per wave
    const int cn0  = blockIdx.x * 64;                // 64 output cols per wave

    v8f acc[4];
    #pragma unroll
    for (int t = 0; t < 4; ++t)
        #pragma unroll
        for (int i = 0; i < 8; ++i) acc[t][i] = 0.0f;

    const unsigned short* arow = xb + (size_t)(wr0 + mrow) * IDIM + half * 8;
    const unsigned short* bcol = Wt + half * 8;

    #pragma unroll 4
    for (int k = 0; k < IDIM; k += 32) {
        v16bf a = ld_frag(arow + k, arow + k + 16);
        #pragma unroll
        for (int t = 0; t < 4; ++t) {
            const unsigned short* bp = bcol + (size_t)(cn0 + t * 16 + mrow) * IDIM + k;
            v16bf bm = ld_frag(bp, bp + 16);
            acc[t] = __builtin_amdgcn_wmma_f32_16x16x32_bf16(
                false, a, false, bm, (short)0, acc[t], false, false);
        }
    }

    const float qscale = (z == 0) ? 0.04419417382415922f : 1.0f;  // 1/sqrt(512)
    #pragma unroll
    for (int t = 0; t < 4; ++t) {
        int col = cn0 + t * 16 + mrow;
        float bc = bias[col];
        #pragma unroll
        for (int i = 0; i < 8; ++i) {
            int row = wr0 + i + half * 8;
            unsigned short hv = f2bf((acc[t][i] + bc) * qscale);
            if (z == 0)      Qs[(size_t)row * DQK + col] = hv;
            else if (z == 1) Ks[(size_t)row * DQK + col] = hv;
            else {
                int bb = row >> 11, s = row & (SEQ - 1);
                Vt[(((size_t)bb * DV + col) << 11) + s] = hv;   // Vt[b][dv][s]
            }
        }
    }
}

// ---------------- flash attention ----------------
// block = 128 threads (4 waves). WG owns 16 query rows; waves split keys for
// scores (16 keys each) and split V-columns (128 each) for the P@V accumulate.
// Q fragments are held register-resident across the whole key loop.
__global__ __launch_bounds__(128) void k_attn(
    const unsigned short* __restrict__ Qs, const unsigned short* __restrict__ Ks,
    const unsigned short* __restrict__ Vt, float* __restrict__ out) {

    __shared__ float          Sbuf[16][64];
    __shared__ unsigned short Pbuf[16][64];

    const int b    = blockIdx.y;
    const int q0   = blockIdx.x * 16;
    const int lane = threadIdx.x & 31;
    const int wave = threadIdx.x >> 5;
    const int mrow = lane & 15;
    const int half = lane >> 4;
    const int vb   = wave * 128;          // this wave's V-column base

    v8f o[8];
    float m[8], l[8];
    #pragma unroll
    for (int i = 0; i < 8; ++i) { m[i] = -1e30f; l[i] = 0.0f; }
    #pragma unroll
    for (int t = 0; t < 8; ++t)
        #pragma unroll
        for (int i = 0; i < 8; ++i) o[t][i] = 0.0f;

    const unsigned short* qrow  = Qs + (size_t)(b * SEQ + q0 + mrow) * DQK + half * 8;
    const unsigned short* kbase = Ks + (size_t)b * SEQ * DQK + half * 8;
    const unsigned short* vbase = Vt + (size_t)b * DV * SEQ;   // [dv][s]

    // Q tile register-resident: 16 fragments x 8 VGPRs = 128 VGPRs.
    v16bf qf[DQK / 32];
    #pragma unroll
    for (int d = 0; d < DQK; d += 32)
        qf[d / 32] = ld_frag(qrow + d, qrow + d + 16);

    for (int kb = 0; kb < SEQ; kb += 64) {
        const int j0 = kb + wave * 16;

        // prefetch next key block's K rows and V tile into cache (global_prefetch_b8)
        if (kb + 64 < SEQ) {
            const unsigned short* knext = kbase + (size_t)(j0 + 64 + mrow) * DQK;
            __builtin_prefetch(knext, 0, 0);
            __builtin_prefetch(knext + 256, 0, 0);
            const unsigned short* vnext =
                vbase + (size_t)(vb + (half * 64) + mrow) * SEQ + kb + 64;
            __builtin_prefetch(vnext, 0, 0);
        }

        // --- 1) score tile: 16 q-rows x 16 keys for this wave ---
        v8f s;
        #pragma unroll
        for (int i = 0; i < 8; ++i) s[i] = 0.0f;
        const unsigned short* kcol = kbase + (size_t)(j0 + mrow) * DQK;
        #pragma unroll
        for (int d = 0; d < DQK; d += 32) {
            v16bf kf = ld_frag(kcol + d, kcol + d + 16);
            s = __builtin_amdgcn_wmma_f32_16x16x32_bf16(
                false, qf[d / 32], false, kf, (short)0, s, false, false);
        }
        // --- 2) stage scores in LDS (scale already folded into Q) ---
        #pragma unroll
        for (int i = 0; i < 8; ++i)
            Sbuf[i + half * 8][wave * 16 + mrow] = s[i];
        __syncthreads();

        // --- 3) online softmax over the 16x64 block (all waves, redundant) ---
        #pragma unroll
        for (int r = 0; r < 8; ++r) {
            int row = r + half * 8;
            float v0 = Sbuf[row][mrow];
            float v1 = Sbuf[row][mrow + 16];
            float v2 = Sbuf[row][mrow + 32];
            float v3 = Sbuf[row][mrow + 48];
            float bm = fmaxf(fmaxf(v0, v1), fmaxf(v2, v3));
            #pragma unroll
            for (int sh = 1; sh < 16; sh <<= 1)
                bm = fmaxf(bm, __shfl_xor(bm, sh, 32));
            float mn   = fmaxf(m[r], bm);
            float corr = __expf(m[r] - mn);
            m[r] = mn;
            float p0 = __expf(v0 - mn), p1 = __expf(v1 - mn);
            float p2 = __expf(v2 - mn), p3 = __expf(v3 - mn);
            if (wave == 0) {
                Pbuf[row][mrow]      = f2bf(p0);
                Pbuf[row][mrow + 16] = f2bf(p1);
                Pbuf[row][mrow + 32] = f2bf(p2);
                Pbuf[row][mrow + 48] = f2bf(p3);
            }
            float rs = p0 + p1 + p2 + p3;
            #pragma unroll
            for (int sh = 1; sh < 16; sh <<= 1)
                rs += __shfl_xor(rs, sh, 32);
            l[r] = l[r] * corr + rs;
            #pragma unroll
            for (int t = 0; t < 8; ++t) o[t][r] *= corr;   // rescale accumulator
        }
        __syncthreads();

        // --- 4) O += P(16x64) @ V(64x128) for this wave's V columns ---
        #pragma unroll
        for (int kk = 0; kk < 64; kk += 32) {
            v16bf pa = ld_frag(&Pbuf[mrow][kk + half * 8],
                               &Pbuf[mrow][kk + half * 8 + 16]);
            #pragma unroll
            for (int t = 0; t < 8; ++t) {
                const unsigned short* vp =
                    vbase + (size_t)(vb + t * 16 + mrow) * SEQ + kb + kk + half * 8;
                v16bf vf = ld_frag(vp, vp + 16);
                o[t] = __builtin_amdgcn_wmma_f32_16x16x32_bf16(
                    false, pa, false, vf, (short)0, o[t], false, false);
            }
        }
    }

    // --- epilogue: normalize and store f32 ---
    float* orow = out + (size_t)(b * SEQ + q0) * DV + vb;
    #pragma unroll
    for (int i = 0; i < 8; ++i) {
        float inv = 1.0f / l[i];
        int row = i + half * 8;
        #pragma unroll
        for (int t = 0; t < 8; ++t)
            orow[(size_t)row * DV + t * 16 + mrow] = o[t][i] * inv;
    }
}

extern "C" void kernel_launch(void* const* d_in, const int* in_sizes, int n_in,
                              void* d_out, int out_size, void* d_ws, size_t ws_size,
                              hipStream_t stream) {
    const float* x  = (const float*)d_in[0];
    const float* Wq = (const float*)d_in[1];
    const float* bq = (const float*)d_in[2];
    const float* Wk = (const float*)d_in[3];
    const float* bk = (const float*)d_in[4];
    const float* Wv = (const float*)d_in[5];
    const float* bv = (const float*)d_in[6];
    float* out = (float*)d_out;

    // workspace layout (bf16 = unsigned short), ~43 MB total
    unsigned short* xb  = (unsigned short*)d_ws;
    unsigned short* Wtq = xb  + (size_t)NROW * IDIM;        // 8192*1024
    unsigned short* Wtk = Wtq + (size_t)DQK * IDIM;
    unsigned short* Wtv = Wtk + (size_t)DQK * IDIM;
    unsigned short* Qs  = Wtv + (size_t)DV  * IDIM;
    unsigned short* Ks  = Qs  + (size_t)NROW * DQK;
    unsigned short* Vt  = Ks  + (size_t)NROW * DQK;

    k_cvt_x<<<dim3((NROW * IDIM) / 256), 256, 0, stream>>>(x, xb, NROW * IDIM);
    k_tr_w <<<dim3((DQK * IDIM) / 256, 3), 256, 0, stream>>>(Wq, Wk, Wv, Wtq, Wtk, Wtv);
    k_proj <<<dim3(DQK / 64, NROW / 128, 3), 256, 0, stream>>>(
        xb, Wtq, Wtk, Wtv, bq, bk, bv, Qs, Ks, Vt);
    k_attn <<<dim3(SEQ / 16, BATCH), 128, 0, stream>>>(Qs, Ks, Vt, out);
}
